// SharedLatentBottleneck_8521215115947
// MI455X (gfx1250) — matrix operand order repaired and checked by reading, not compiled, not run on hardware
//
#include <hip/hip_runtime.h>
#include <hip/hip_bf16.h>

// ---------------------------------------------------------------------------
// SharedLatentBottleneck for MI455X (gfx1250, wave32, WMMA bf16 16x16x32).
// GEMMs: compile-time shapes, weights staged fragment-ordered in LDS as bf16,
// activations converted f32->bf16 with native v_cvt_pk_bf16_f32, f32
// accumulation, two 16-row tiles per wave sharing each B fragment (tail
// handled by a separate RT=1 instantiation so the hot path is branch-free).
// Attention: 3-pass atomic segment softmax (u32-mapped atomicMax, f32 atomicAdd).
// ---------------------------------------------------------------------------

typedef __attribute__((ext_vector_type(16))) __bf16          v16bf;
typedef __attribute__((ext_vector_type(8)))  float           v8f;

static constexpr int  NN = 50000;
static constexpr int  EE = 400000;
static constexpr int  SS = NN + EE;     // 450000
static constexpr int  DNF = 256;        // node feature dim
static constexpr int  DEF = 128;        // edge feature dim
static constexpr int  DLA = 128;        // latent dim
static constexpr int  HH  = 4;          // heads

__device__ __forceinline__ unsigned mapf(float f) {
    unsigned u = __float_as_uint(f);
    return (u & 0x80000000u) ? ~u : (u | 0x80000000u);   // order-preserving map
}
__device__ __forceinline__ float unmapf(unsigned u) {
    unsigned b = (u & 0x80000000u) ? (u & 0x7FFFFFFFu) : ~u;
    return __uint_as_float(b);
}

__device__ __forceinline__ float wave_sum(float v) {
    #pragma unroll
    for (int m = 1; m < 32; m <<= 1) v += __shfl_xor(v, m, 32);
    return v;
}

// src/tgt for cross-edge c in [0, 4E): [src,eidx | tgt,eidx | eidx,src | eidx,tgt]
__device__ __forceinline__ void cross_edge(long long c, const int* __restrict__ ei,
                                           int& src, int& tgt) {
    int blk      = (int)(c / EE);
    long long e  = c - (long long)blk * EE;
    int x        = (blk & 1) ? ei[EE + e] : ei[e];
    int eidx     = (int)(NN + e);
    if (blk < 2) { src = x;    tgt = eidx; }
    else         { src = eidx; tgt = x;    }
}

// ---------------------------------------------------------------------------
// GEMM super-tile body: RT row tiles (16 rows each) x CT column tiles per
// wave, B fragments from LDS reused across the RT row tiles. Straight-line,
// wave-uniform control flow only (EXEC all-ones around every WMMA).
// ---------------------------------------------------------------------------
template<int Kd, int Nc, int RT>
__device__ __forceinline__
void gemm_supertile(const float* __restrict__ A, float* __restrict__ C,
                    const v16bf* __restrict__ sBf, const float* __restrict__ sBias,
                    int rt, int wave, int lane)
{
    constexpr int KT = Kd / 32;
    constexpr int NT = Nc / 16;
    constexpr int CT = (NT + 7) / 8;

    const int r0  = lane & 15;
    const int kbA = (lane >> 4) << 3;
    const int rOutOfs = (lane >> 4) << 3;

    const float* Arow[RT];
    #pragma unroll
    for (int rr = 0; rr < RT; ++rr)
        Arow[rr] = A + (size_t)((rt + rr) * 16 + r0) * Kd;

    v8f acc[CT][RT];
    #pragma unroll
    for (int c = 0; c < CT; ++c) {
        const float bval = sBias[(wave + 8 * c) * 16 + r0];
        #pragma unroll
        for (int rr = 0; rr < RT; ++rr)
            #pragma unroll
            for (int r = 0; r < 8; ++r) acc[c][rr][r] = bval;
    }

    #pragma unroll
    for (int kt = 0; kt < KT; ++kt) {
        const int k0 = kt * 32;
        v16bf a[RT];
        #pragma unroll
        for (int rr = 0; rr < RT; ++rr) {
            const float* ap = Arow[rr] + k0 + kbA;
            #pragma unroll
            for (int e = 0; e < 8; ++e) {
                a[rr][e]     = (__bf16)ap[e];
                a[rr][8 + e] = (__bf16)ap[16 + e];
            }
        }
        #pragma unroll
        for (int c = 0; c < CT; ++c) {
            const int ct = wave + 8 * c;
            const v16bf bfrag = sBf[(ct * KT + kt) * 32 + lane];
            #pragma unroll
            for (int rr = 0; rr < RT; ++rr)
                acc[c][rr] = __builtin_amdgcn_wmma_f32_16x16x32_bf16(
                                 false, a[rr], false, bfrag, (short)0,
                                 acc[c][rr], false, false);
        }
    }

    #pragma unroll
    for (int c = 0; c < CT; ++c) {
        const int col = (wave + 8 * c) * 16 + r0;
        #pragma unroll
        for (int rr = 0; rr < RT; ++rr) {
            float* Cp = C + (size_t)((rt + rr) * 16 + rOutOfs) * Nc + col;
            #pragma unroll
            for (int r = 0; r < 8; ++r) Cp[(size_t)r * Nc] = acc[c][rr][r];
        }
    }
}

// ---------------------------------------------------------------------------
// C[M,Nc] = A[M,Kd] @ B[Kd,Nc] + bias   (f32 in memory, bf16 WMMA compute)
// Block = 256 threads = 8 waves. B staged in LDS fragment-ordered:
//   sB[((ct*KT + kt)*32 + laneId)*16 + e]  -> 32 contiguous bytes per lane.
// ---------------------------------------------------------------------------
extern __shared__ char smem_raw[];

template<int Kd, int Nc>
__global__ __launch_bounds__(256)
void wmma_gemm_bias_kernel(const float* __restrict__ A, const float* __restrict__ B,
                           const float* __restrict__ bias, float* __restrict__ C,
                           int M)
{
    constexpr int KT = Kd / 32;

    __bf16* sB    = (__bf16*)smem_raw;                         // [NT*KT*32*16]
    float*  sBias = (float*)(smem_raw + (size_t)Kd * Nc * 2);  // [Nc]

    const int tid  = threadIdx.x;
    const int wave = tid >> 5;
    const int lane = tid & 31;

    // Stage B swizzled into fragment order, converting with native bf16 cvt.
    for (int i = tid; i < Kd * Nc; i += 256) {
        const int k  = i / Nc, n = i % Nc;
        const int ct = n >> 4, cn = n & 15;
        const int kt = k >> 5, kr = k & 31;
        const int half = (kr >> 3) & 1, e = (kr & 7) | ((kr >> 4) << 3);
        sB[(size_t)(((ct * KT + kt) * 32) + half * 16 + cn) * 16 + e] = (__bf16)B[i];
    }
    for (int i = tid; i < Nc; i += 256) sBias[i] = bias[i];
    __syncthreads();

    const v16bf* sBf = (const v16bf*)sB;

    const int MT  = M >> 4;
    const int MT2 = (MT + 1) >> 1;       // 32-row super tiles (tail via RT=1)

    for (int rt2 = blockIdx.x; rt2 < MT2; rt2 += gridDim.x) {
        const int rt = rt2 * 2;
        if (rt2 + (int)gridDim.x < MT2)
            __builtin_prefetch(A + (size_t)((rt2 + gridDim.x) * 32 + (lane & 15)) * Kd, 0, 0);
        if (rt + 1 < MT)
            gemm_supertile<Kd, Nc, 2>(A, C, sBf, sBias, rt, wave, lane);
        else
            gemm_supertile<Kd, Nc, 1>(A, C, sBf, sBias, rt, wave, lane);
    }
}

// ---------------------------------------------------------------------------
// Pass 1: per-cross-edge per-head scores + atomic segment max.
// One wave per cross-edge; lane l covers dims [4l,4l+4), head = l>>3.
// ---------------------------------------------------------------------------
__global__ __launch_bounds__(256)
void attn_scores_kernel(const float* __restrict__ Q, const float* __restrict__ Km,
                        const int* __restrict__ ei, float* __restrict__ scores,
                        unsigned* __restrict__ maxv)
{
    long long wid = ((long long)blockIdx.x * blockDim.x + threadIdx.x) >> 5;
    const int lane = threadIdx.x & 31;
    if (wid >= (long long)4 * EE) return;

    int src, tgt;
    cross_edge(wid, ei, src, tgt);

    const float4 q = ((const float4*)(Q  + (size_t)tgt * DLA))[lane];
    const float4 k = ((const float4*)(Km + (size_t)src * DLA))[lane];
    float p = q.x * k.x + q.y * k.y + q.z * k.z + q.w * k.w;
    p += __shfl_xor(p, 1, 32);
    p += __shfl_xor(p, 2, 32);
    p += __shfl_xor(p, 4, 32);           // reduce within 8-lane head group

    const int h = lane >> 3;
    if ((lane & 7) == 0) {
        const float s = p * 0.17677669529663687f;   // 1/sqrt(32)
        scores[wid * HH + h] = s;
        atomicMax(maxv + (size_t)tgt * HH + h, mapf(s));
    }
}

// ---------------------------------------------------------------------------
// Pass 2: exp(s - max) in place + atomic segment sum. Thread per (c,h).
// ---------------------------------------------------------------------------
__global__ __launch_bounds__(256)
void attn_exp_kernel(float* __restrict__ scores, const unsigned* __restrict__ maxv,
                     float* __restrict__ sumexp, const int* __restrict__ ei)
{
    long long i = (long long)blockIdx.x * blockDim.x + threadIdx.x;
    if (i >= (long long)4 * EE * HH) return;
    long long c = i >> 2;
    int h = (int)(i & 3);

    int src, tgt;
    cross_edge(c, ei, src, tgt);
    (void)src;

    const float m  = unmapf(maxv[(size_t)tgt * HH + h]);
    const float ex = __expf(scores[i] - m);
    scores[i] = ex;
    atomicAdd(sumexp + (size_t)tgt * HH + h, ex);
}

// ---------------------------------------------------------------------------
// Pass 3: out[tgt] += attn * V[src]. One wave per cross-edge, float4 gather,
// 4 scalar f32 scatter-adds per lane.
// ---------------------------------------------------------------------------
__global__ __launch_bounds__(256)
void attn_apply_kernel(const float* __restrict__ V, const int* __restrict__ ei,
                       const float* __restrict__ scores, const float* __restrict__ sumexp,
                       float* __restrict__ outacc)
{
    long long wid = ((long long)blockIdx.x * blockDim.x + threadIdx.x) >> 5;
    const int lane = threadIdx.x & 31;
    if (wid >= (long long)4 * EE) return;

    int src, tgt;
    cross_edge(wid, ei, src, tgt);

    const int h = lane >> 3;
    const float w = scores[wid * HH + h] /
                    (sumexp[(size_t)tgt * HH + h] + 1e-10f);

    const float4 v = ((const float4*)(V + (size_t)src * DLA))[lane];
    float* o = outacc + (size_t)tgt * DLA + lane * 4;
    atomicAdd(o + 0, w * v.x);
    atomicAdd(o + 1, w * v.y);
    atomicAdd(o + 2, w * v.z);
    atomicAdd(o + 3, w * v.w);
}

// ---------------------------------------------------------------------------
// y = LayerNorm(base + g*upd) * gamma + beta, g = sigmoid(*gate) (1 if null).
// One wave per row, D in {128, 256}, grid-stride over rows.
// ---------------------------------------------------------------------------
__global__ __launch_bounds__(256)
void residual_ln_kernel(const float* __restrict__ base, const float* __restrict__ upd,
                        const float* __restrict__ gate,
                        const float* __restrict__ gamma, const float* __restrict__ beta,
                        float* __restrict__ out, long long rows, int D)
{
    long long wid    = ((long long)blockIdx.x * blockDim.x + threadIdx.x) >> 5;
    long long stride = ((long long)gridDim.x * blockDim.x) >> 5;
    const int lane   = threadIdx.x & 31;
    const int chunks = D >> 7;                         // float4s per lane (1 or 2)
    const float g    = gate ? 1.0f / (1.0f + __expf(-gate[0])) : 1.0f;
    const float invD = 1.0f / (float)D;

    for (long long r = wid; r < rows; r += stride) {
        float4 x[2];
        float s = 0.0f;
        #pragma unroll
        for (int ci = 0; ci < 2; ++ci) {
            if (ci >= chunks) break;
            const int d = (ci << 7) + lane * 4;
            const float4 b = *(const float4*)(base + (size_t)r * D + d);
            const float4 u = *(const float4*)(upd  + (size_t)r * D + d);
            x[ci].x = b.x + g * u.x;  x[ci].y = b.y + g * u.y;
            x[ci].z = b.z + g * u.z;  x[ci].w = b.w + g * u.w;
            s += x[ci].x + x[ci].y + x[ci].z + x[ci].w;
        }
        const float mean = wave_sum(s) * invD;
        float vs = 0.0f;
        #pragma unroll
        for (int ci = 0; ci < 2; ++ci) {
            if (ci >= chunks) break;
            const float dx0 = x[ci].x - mean, dx1 = x[ci].y - mean;
            const float dx2 = x[ci].z - mean, dx3 = x[ci].w - mean;
            vs += dx0 * dx0 + dx1 * dx1 + dx2 * dx2 + dx3 * dx3;
        }
        const float inv = rsqrtf(wave_sum(vs) * invD + 1e-5f);
        #pragma unroll
        for (int ci = 0; ci < 2; ++ci) {
            if (ci >= chunks) break;
            const int d = (ci << 7) + lane * 4;
            const float4 gm = *(const float4*)(gamma + d);
            const float4 bt = *(const float4*)(beta  + d);
            float4 y;
            y.x = (x[ci].x - mean) * inv * gm.x + bt.x;
            y.y = (x[ci].y - mean) * inv * gm.y + bt.y;
            y.z = (x[ci].z - mean) * inv * gm.z + bt.z;
            y.w = (x[ci].w - mean) * inv * gm.w + bt.w;
            *(float4*)(out + (size_t)r * D + d) = y;
        }
    }
}

// ---------------------------------------------------------------------------
extern "C" void kernel_launch(void* const* d_in, const int* in_sizes, int n_in,
                              void* d_out, int out_size, void* d_ws, size_t ws_size,
                              hipStream_t stream)
{
    const float* node_features = (const float*)d_in[0];
    const float* edge_features = (const float*)d_in[1];
    const int*   edge_index    = (const int*)d_in[2];
    const float* Wnd = (const float*)d_in[3];  const float* bnd = (const float*)d_in[4];
    const float* Wed = (const float*)d_in[5];  const float* bed = (const float*)d_in[6];
    const float* Wq  = (const float*)d_in[7];  const float* bq  = (const float*)d_in[8];
    const float* Wk  = (const float*)d_in[9];  const float* bk  = (const float*)d_in[10];
    const float* Wv  = (const float*)d_in[11]; const float* bv  = (const float*)d_in[12];
    const float* Wo  = (const float*)d_in[13]; const float* bo  = (const float*)d_in[14];
    const float* gs  = (const float*)d_in[15]; const float* bs  = (const float*)d_in[16];
    const float* Wnu = (const float*)d_in[17]; const float* bnu = (const float*)d_in[18];
    const float* Weu = (const float*)d_in[19]; const float* beu = (const float*)d_in[20];
    const float* node_gate = (const float*)d_in[21];
    const float* edge_gate = (const float*)d_in[22];
    const float* gn  = (const float*)d_in[23]; const float* bn  = (const float*)d_in[24];
    const float* ge  = (const float*)d_in[25]; const float* be  = (const float*)d_in[26];
    float* out = (float*)d_out;

    // Workspace layout (floats)
    const size_t SDL = (size_t)SS * DLA;
    float* ws       = (float*)d_ws;
    float* combined = ws;                                   // [S,128]
    float* Qb       = combined + SDL;                       // [S,128]
    float* Kb       = Qb + SDL;                             // [S,128]
    float* Vb       = Kb + SDL;                             // [S,128]
    float* attn     = Vb + SDL;                             // [S,128] attn out / edge_upd
    float* proj     = attn + SDL;                           // [S,128] proj / node_upd
    float* scores   = proj + SDL;                           // [4E,4]
    float* sumexp   = scores + (size_t)4 * EE * HH;         // [S,4]
    unsigned* maxv  = (unsigned*)(sumexp + (size_t)SS * HH);// [S,4]

    (void)hipMemsetAsync(attn,   0, SDL * sizeof(float), stream);
    (void)hipMemsetAsync(sumexp, 0, (size_t)SS * HH * sizeof(float), stream);
    (void)hipMemsetAsync(maxv,   0, (size_t)SS * HH * sizeof(unsigned), stream);

    auto gridFor = [](int M) {
        const int MT2 = ((M >> 4) + 1) >> 1;
        return MT2 < 2048 ? MT2 : 2048;
    };
    auto shFor = [](int Kd, int Nc) {
        return (size_t)Kd * Nc * 2 + (size_t)Nc * 4;
    };

    // Encoders -> combined latent
    wmma_gemm_bias_kernel<256,128><<<gridFor(NN), 256, shFor(256,128), stream>>>(
        node_features, Wnd, bnd, combined, NN);
    wmma_gemm_bias_kernel<128,128><<<gridFor(EE), 256, shFor(128,128), stream>>>(
        edge_features, Wed, bed, combined + (size_t)NN * DLA, EE);

    // Q / K / V
    wmma_gemm_bias_kernel<128,128><<<gridFor(SS), 256, shFor(128,128), stream>>>(
        combined, Wq, bq, Qb, SS);
    wmma_gemm_bias_kernel<128,128><<<gridFor(SS), 256, shFor(128,128), stream>>>(
        combined, Wk, bk, Kb, SS);
    wmma_gemm_bias_kernel<128,128><<<gridFor(SS), 256, shFor(128,128), stream>>>(
        combined, Wv, bv, Vb, SS);

    // Segment-softmax attention over 4E cross edges (wave per edge)
    const long long waves4E  = (long long)4 * EE;
    const int blocks_edge    = (int)((waves4E + 7) / 8);          // 8 waves/block
    const int blocks_exp     = (int)((waves4E * HH + 255) / 256);
    attn_scores_kernel<<<blocks_edge, 256, 0, stream>>>(Qb, Kb, edge_index, scores, maxv);
    attn_exp_kernel   <<<blocks_exp,  256, 0, stream>>>(scores, maxv, sumexp, edge_index);
    attn_apply_kernel <<<blocks_edge, 256, 0, stream>>>(Vb, edge_index, scores, sumexp, attn);

    // Output projection + mid layernorm (in place on combined)
    wmma_gemm_bias_kernel<128,128><<<gridFor(SS), 256, shFor(128,128), stream>>>(
        attn, Wo, bo, proj, SS);
    residual_ln_kernel<<<4096, 256, 0, stream>>>(combined, proj, nullptr, gs, bs,
                                                 combined, (long long)SS, DLA);

    // Node decoder + final node layernorm -> out[0 : N*256)
    wmma_gemm_bias_kernel<128,256><<<gridFor(NN), 256, shFor(128,256), stream>>>(
        combined, Wnu, bnu, proj, NN);
    residual_ln_kernel<<<4096, 256, 0, stream>>>(node_features, proj, node_gate, gn, bn,
                                                 out, (long long)NN, DNF);

    // Edge decoder + final edge layernorm -> out[N*256 : ...)
    wmma_gemm_bias_kernel<128,128><<<gridFor(EE), 256, shFor(128,128), stream>>>(
        combined + (size_t)NN * DLA, Weu, beu, attn, EE);
    residual_ln_kernel<<<4096, 256, 0, stream>>>(edge_features, attn, edge_gate, ge, be,
                                                 out + (size_t)NN * DNF, (long long)EE, DEF);
}
